// SimpleFNO_33079838114334
// MI455X (gfx1250) — matrix operand order repaired and checked by reading, not compiled, not run on hardware
//
#include <hip/hip_runtime.h>
#include <hip/hip_bf16.h>
#include <math.h>
#include <stdint.h>

// Problem constants (from reference)
#define B_      128
#define L_      2048
#define IND     8
#define C_      256
#define MODES_  16
#define NF      32      // 2*MODES (real | imag columns)
#define FC1N    128

typedef __bf16 bf16;
typedef __attribute__((ext_vector_type(16))) __bf16 v16bf;
typedef __attribute__((ext_vector_type(8)))  float  v8f;
typedef __attribute__((ext_vector_type(4)))  int    v4i_;

union FragU { v16bf v; uint4 q[2]; };

// CDNA5 async global->LDS path (guarded so the build cannot break)
#if defined(__gfx1250__) && \
    __has_builtin(__builtin_amdgcn_global_load_async_to_lds_b128) && \
    __has_builtin(__builtin_amdgcn_s_wait_asynccnt)
#define HAVE_ASYNC_LDS 1
#define TO_GLOBAL(p) ((__attribute__((address_space(1))) v4i_*)(uintptr_t)(p))
#define TO_LDS(p)    ((__attribute__((address_space(3))) v4i_*)(uintptr_t)(p))
#else
#define HAVE_ASYNC_LDS 0
#endif

// ---------------------------------------------------------------------------
// WMMA helpers (wave32; 16x16x32 bf16 -> f32 accumulate)
// ---------------------------------------------------------------------------
__device__ __forceinline__ v8f wmma_bf16(v16bf a, v16bf b, v8f c) {
  return __builtin_amdgcn_wmma_f32_16x16x32_bf16(
      /*neg_a=*/false, a, /*neg_b=*/false, b,
      /*c_mod=*/(short)0, c, /*reuse_a=*/false, /*reuse_b=*/false);
}

// A fragment: 16x32 bf16, row-major (rows = M, contiguous K).
// Lane layout: lane = 16*half + m ; elements 0..7 = K[half*8 .. +7],
//              elements 8..15 = K[16+half*8 .. +7]   (two 16B loads)
__device__ __forceinline__ v16bf loadA(const bf16* base, int ld) {
  const int lane = threadIdx.x & 31;
  const int m = lane & 15, hh = lane >> 4;
  const bf16* p = base + (long)m * ld + hh * 8;
  FragU f;
  f.q[0] = *(const uint4*)(p);
  f.q[1] = *(const uint4*)(p + 16);
  return f.v;
}

// B fragment: 32x16 bf16 where memory is "n-major": elem(K,n) = base[n*ld + K].
// Lane layout: lane = 16*half + n ; elements 0..7 = K[half*8 .. +7],
//              elements 8..15 = K[16+half*8 .. +7]
__device__ __forceinline__ v16bf loadB_nmajor(const bf16* base, int ld) {
  const int lane = threadIdx.x & 31;
  const int n = lane & 15, hh = lane >> 4;
  const bf16* p = base + (long)n * ld + hh * 8;
  FragU f;
  f.q[0] = *(const uint4*)(p);
  f.q[1] = *(const uint4*)(p + 16);
  return f.v;
}

__device__ __forceinline__ float gelu_exact(float x) {
  return 0.5f * x * (1.0f + erff(x * 0.70710678118654752f));
}

// ---------------------------------------------------------------------------
// fc0: h[b,l,c] = x[b,l,:8] @ fc0_w + b ; written in two layouts (bf16):
//   h_cl[b][c][l]  (DFT A-fragments: contiguous K=l)
//   h_lc[b][l][c]  (pointwise B-fragments: n-major, contiguous K=i)
// ---------------------------------------------------------------------------
__global__ void k_fc0(const float* __restrict__ x, const float* __restrict__ w,
                      const float* __restrict__ bias,
                      bf16* __restrict__ h_cl, bf16* __restrict__ h_lc) {
  const int idx = blockIdx.x * blockDim.x + threadIdx.x; // (b,l,c), c fastest
  const int c = idx & (C_ - 1);
  const int l = (idx >> 8) & (L_ - 1);
  const int b = idx >> 19;
  const float* xp = x + (long)(b * L_ + l) * IND;
  float acc = bias[c];
#pragma unroll
  for (int i = 0; i < IND; ++i) acc += xp[i] * w[i * C_ + c];
  const bf16 hv = (bf16)acc;
  h_lc[((long)b * L_ + l) * C_ + c] = hv;
  h_cl[((long)b * C_ + c) * L_ + l] = hv;
}

// ---------------------------------------------------------------------------
// DFT bases (bf16):
//   Ft[f][l], f<16: cos(2*pi*f*l/L) ; f>=16: -sin(2*pi*(f-16)*l/L)
//   Glm[l][k], k<16: a_k*cos ; k>=16: -a_k*sin (a_0=1/L else 2/L; imag(X0) dropped)
// ---------------------------------------------------------------------------
__global__ void k_basis(bf16* __restrict__ Ft, bf16* __restrict__ Glm) {
  const int l = blockIdx.x * blockDim.x + threadIdx.x;
  if (l >= L_) return;
  const float w0 = 6.28318530717958647692f / (float)L_;
  for (int k = 0; k < MODES_; ++k) {
    const float ang = w0 * (float)k * (float)l;
    const float cv = cosf(ang), sv = sinf(ang);
    Ft[(long)k * L_ + l]            = (bf16)cv;
    Ft[(long)(k + MODES_) * L_ + l] = (bf16)(-sv);
    const float a = (k == 0) ? (1.0f / L_) : (2.0f / L_);
    Glm[(long)l * NF + k]          = (bf16)(a * cv);
    Glm[(long)l * NF + k + MODES_] = (bf16)((k == 0) ? 0.0f : (-a * sv));
  }
}

// Dense weight fp32 -> bf16 (same [o][i] layout: A-matrix row-major)
__global__ void k_cvt_dense(const float* __restrict__ in, bf16* __restrict__ out) {
  const int idx = blockIdx.x * blockDim.x + threadIdx.x;
  out[idx] = (bf16)in[idx];
}

// Spectral weights (i,o,k) fp32 -> W[k][o][i] bf16 (B-fragment n-major: n=o, K=i)
__global__ void k_cvt_spec(const float* __restrict__ in, bf16* __restrict__ out) {
  const int idx = blockIdx.x * blockDim.x + threadIdx.x; // (k,o,i), i fastest
  const int i = idx & 255;
  const int o = (idx >> 8) & 255;
  const int k = idx >> 16;
  out[(((long)k * C_ + o) * C_) + i] = (bf16)in[((long)i * C_ + o) * MODES_ + k];
}

// ---------------------------------------------------------------------------
// Forward DFT as GEMM: Xf[m, f] = sum_l h_cl[m, l] * Ft[f, l]
//   Each wave: 2 row tiles x 2 freq tiles = 4 independent WMMA chains,
//   reusing both A and B fragments. K = 2048 (64 k-steps).
// ---------------------------------------------------------------------------
__global__ void k_dft(const bf16* __restrict__ h_cl, const bf16* __restrict__ Ft,
                      bf16* __restrict__ Xr, bf16* __restrict__ Xi) {
  const int wave = blockIdx.x * (blockDim.x >> 5) + (threadIdx.x >> 5);
  const bf16* A0 = h_cl + (long)(2 * wave) * 16 * L_;
  const bf16* A1 = A0 + (long)16 * L_;
  v8f acc00 = {}, acc01 = {}, acc10 = {}, acc11 = {};
  for (int kk = 0; kk < L_ / 32; ++kk) {
    __builtin_prefetch(A0 + (kk + 2) * 32, 0, 1);
    __builtin_prefetch(A1 + (kk + 2) * 32, 0, 1);
    const v16bf a0 = loadA(A0 + kk * 32, L_);
    const v16bf a1 = loadA(A1 + kk * 32, L_);
    const v16bf b0 = loadB_nmajor(Ft + kk * 32, L_);
    const v16bf b1 = loadB_nmajor(Ft + (long)MODES_ * L_ + kk * 32, L_);
    acc00 = wmma_bf16(a0, b0, acc00);
    acc01 = wmma_bf16(a0, b1, acc01);
    acc10 = wmma_bf16(a1, b0, acc10);
    acc11 = wmma_bf16(a1, b1, acc11);
  }
  const int lane = threadIdx.x & 31;
  const int n = lane & 15, hh = lane >> 4;
#pragma unroll
  for (int r = 0; r < 8; ++r) {
    const int m0 = 2 * wave * 16 + r + 8 * hh;
    const int m1 = m0 + 16;
    const int b0i = m0 >> 8, i0 = m0 & 255;
    const int b1i = m1 >> 8, i1 = m1 & 255;
    Xr[((long)n * B_ + b0i) * C_ + i0] = (bf16)acc00[r];
    Xi[((long)n * B_ + b0i) * C_ + i0] = (bf16)acc01[r];
    Xr[((long)n * B_ + b1i) * C_ + i1] = (bf16)acc10[r];
    Xi[((long)n * B_ + b1i) * C_ + i1] = (bf16)acc11[r];
  }
}

// ---------------------------------------------------------------------------
// Per-mode complex mix: Y = X_k @ W_k (complex), M=128(b) N=256(o) K=256(i)
//   Yr = Xr@Wr - Xi@Wi ; Yi = Xr@Wi + Xi@Wr  (4 independent WMMA chains)
//   Output packed for iDFT: Y[(b*C+o)*32 + k] = Yr, [+16+k] = Yi (bf16)
// ---------------------------------------------------------------------------
__global__ void k_modemix(const bf16* __restrict__ Xr, const bf16* __restrict__ Xi,
                          const bf16* __restrict__ Wr, const bf16* __restrict__ Wi,
                          bf16* __restrict__ Y) {
  const int wave = blockIdx.x * (blockDim.x >> 5) + (threadIdx.x >> 5);
  const int co = wave & 15;        // o tile
  const int rb = (wave >> 4) & 7;  // b tile
  const int k  = wave >> 7;        // mode
  const bf16* Ar = Xr + ((long)k * B_ + rb * 16) * C_;
  const bf16* Ai = Xi + ((long)k * B_ + rb * 16) * C_;
  const bf16* Br = Wr + ((long)k * C_ + co * 16) * C_; // [k][o][i], n-major
  const bf16* Bi = Wi + ((long)k * C_ + co * 16) * C_;
  v8f arr = {}, aii = {}, ari = {}, air = {};
  for (int kk = 0; kk < C_ / 32; ++kk) {
    const v16bf far_ = loadA(Ar + kk * 32, C_);
    const v16bf fai_ = loadA(Ai + kk * 32, C_);
    const v16bf fbr_ = loadB_nmajor(Br + kk * 32, C_);
    const v16bf fbi_ = loadB_nmajor(Bi + kk * 32, C_);
    arr = wmma_bf16(far_, fbr_, arr);
    aii = wmma_bf16(fai_, fbi_, aii);
    ari = wmma_bf16(far_, fbi_, ari);
    air = wmma_bf16(fai_, fbr_, air);
  }
  const int lane = threadIdx.x & 31;
  const int n = lane & 15, hh = lane >> 4;
#pragma unroll
  for (int r = 0; r < 8; ++r) {
    const int b = rb * 16 + r + 8 * hh;
    const int o = co * 16 + n;
    const long row = ((long)b * C_ + o) * NF;
    Y[row + k]          = (bf16)(arr[r] - aii[r]);
    Y[row + MODES_ + k] = (bf16)(ari[r] + air[r]);
  }
}

// ---------------------------------------------------------------------------
// Fused FNO block tile: out = gelu( iDFT(Y) + W @ h + bias )
//   The 16x256 weight strip (shared by all waves of the block) is staged into
//   LDS via the CDNA5 async global->LDS path; A-fragments then come from LDS
//   (ds_load_b128). Each wave owns NT independent 16x16 accumulators (shared
//   A fragment per K-step) for WMMA latency hiding.
// ---------------------------------------------------------------------------
template <int NWAVES, int NT>
__global__ void k_fno_block(const bf16* __restrict__ h_lc, int h_rows, int h_off,
                            const bf16* __restrict__ Y,
                            const bf16* __restrict__ W,
                            const float* __restrict__ bias,
                            const bf16* __restrict__ Glm,
                            int l_base,
                            bf16* __restrict__ out_cl,
                            bf16* __restrict__ out_tail,
                            float* __restrict__ out_last) {
  __shared__ __align__(16) bf16 sW[16 * C_]; // 8 KB weight strip [o_local][i]
  const int b  = blockIdx.x;
  const int ot = blockIdx.y;
  const int wv = threadIdx.x >> 5;
  const int l0 = l_base + (blockIdx.z * NWAVES + wv) * (NT * 16);

  // --- stage weight strip into LDS ---
  {
    const bf16* gW = W + (long)ot * 16 * C_;
#if HAVE_ASYNC_LDS
#pragma unroll
    for (int ofs = 0; ofs < 16 * C_; ofs += NWAVES * 32 * 8) {
      const int e = ofs + threadIdx.x * 8; // 8 bf16 = 16B per lane
      __builtin_amdgcn_global_load_async_to_lds_b128(
          TO_GLOBAL(gW + e), TO_LDS(sW + e), 0, 0);
    }
    __builtin_amdgcn_s_wait_asynccnt(0);
    __syncthreads();
#else
#pragma unroll
    for (int ofs = 0; ofs < 16 * C_; ofs += NWAVES * 32 * 8) {
      const int e = ofs + threadIdx.x * 8;
      *(uint4*)(sW + e) = *(const uint4*)(gW + e);
    }
    __syncthreads();
#endif
  }

  v8f acc[NT] = {};
  // --- spectral (inverse DFT) contribution: single K=32 step per tile ---
  {
    const v16bf a = loadA(Y + ((long)b * C_ + ot * 16) * NF, NF);
#pragma unroll
    for (int t = 0; t < NT; ++t) {
      const v16bf bb = loadB_nmajor(Glm + (long)(l0 + t * 16) * NF, NF);
      acc[t] = wmma_bf16(a, bb, acc[t]);
    }
  }
  // --- pointwise conv: A from LDS, B streamed from h_lc ---
  const bf16* Bbase = h_lc + ((long)b * h_rows + (l0 - h_off)) * C_;
  for (int kk = 0; kk < C_ / 32; ++kk) {
    const v16bf a = loadA(sW + kk * 32, C_);
#pragma unroll
    for (int t = 0; t < NT; ++t) {
      __builtin_prefetch(Bbase + (long)t * 16 * C_ + (kk + 1) * 32, 0, 1);
      const v16bf bb = loadB_nmajor(Bbase + (long)t * 16 * C_ + kk * 32, C_);
      acc[t] = wmma_bf16(a, bb, acc[t]);
    }
  }
  // --- bias + exact GELU + stores ---
  const int lane = threadIdx.x & 31;
  const int n = lane & 15, hh = lane >> 4;
#pragma unroll
  for (int t = 0; t < NT; ++t) {
    const int l = l0 + t * 16 + n;
#pragma unroll
    for (int r = 0; r < 8; ++r) {
      const int o = ot * 16 + r + 8 * hh;
      float v = acc[t][r] + bias[o];
      v = gelu_exact(v);
      if (out_cl)   out_cl[((long)b * C_ + o) * L_ + l] = (bf16)v;
      if (out_tail && l >= L_ - 16)
        out_tail[((long)b * 16 + (l - (L_ - 16))) * C_ + o] = (bf16)v;
      if (out_last && l == L_ - 1) out_last[(long)b * C_ + o] = v;
    }
  }
}

// ---------------------------------------------------------------------------
// Head: g[b,:256] -> gelu(fc1) -> fc2 ; one block per batch, LDS reduction.
// ---------------------------------------------------------------------------
__global__ void k_head(const float* __restrict__ g, const float* __restrict__ w1,
                       const float* __restrict__ b1, const float* __restrict__ w2,
                       const float* __restrict__ b2, float* __restrict__ out) {
  __shared__ float red[FC1N];
  const int b = blockIdx.x, j = threadIdx.x;
  const float* gb = g + (long)b * C_;
  float acc = b1[j];
  for (int c = 0; c < C_; ++c) acc += gb[c] * w1[c * FC1N + j];
  acc = gelu_exact(acc);
  red[j] = acc * w2[j];
  __syncthreads();
  for (int s = FC1N / 2; s > 0; s >>= 1) {
    if (j < s) red[j] += red[j + s];
    __syncthreads();
  }
  if (j == 0) out[b] = red[0] + b2[0];
}

// ---------------------------------------------------------------------------
extern "C" void kernel_launch(void* const* d_in, const int* in_sizes, int n_in,
                              void* d_out, int out_size, void* d_ws, size_t ws_size,
                              hipStream_t stream) {
  (void)in_sizes; (void)n_in; (void)out_size; (void)ws_size;
  const float* x     = (const float*)d_in[0];
  const float* fc0_w = (const float*)d_in[1];
  const float* fc0_b = (const float*)d_in[2];
  const float* sw0_r = (const float*)d_in[3];
  const float* sw0_i = (const float*)d_in[4];
  const float* w0_w  = (const float*)d_in[5];
  const float* w0_b  = (const float*)d_in[6];
  const float* sw1_r = (const float*)d_in[7];
  const float* sw1_i = (const float*)d_in[8];
  const float* w1_w  = (const float*)d_in[9];
  const float* w1_b  = (const float*)d_in[10];
  const float* fc1_w = (const float*)d_in[11];
  const float* fc1_b = (const float*)d_in[12];
  const float* fc2_w = (const float*)d_in[13];
  const float* fc2_b = (const float*)d_in[14];
  float* out = (float*)d_out;

  char* ws = (char*)d_ws;
  size_t off = 0;
  auto carve = [&](size_t bytes) -> char* {
    char* p = ws + off;
    off += (bytes + 255) & ~(size_t)255;
    return p;
  };
  bf16* h_cl    = (bf16*)carve((size_t)B_ * C_ * L_ * 2);      // 128 MB
  bf16* h_lc    = (bf16*)carve((size_t)B_ * L_ * C_ * 2);      // 128 MB
  bf16* h1_cl   = (bf16*)carve((size_t)B_ * C_ * L_ * 2);      // 128 MB
  bf16* h1_tail = (bf16*)carve((size_t)B_ * 16 * C_ * 2);      // 1 MB
  bf16* Ft      = (bf16*)carve((size_t)NF * L_ * 2);
  bf16* Glm     = (bf16*)carve((size_t)L_ * NF * 2);
  bf16* w0bf    = (bf16*)carve((size_t)C_ * C_ * 2);
  bf16* w1bf    = (bf16*)carve((size_t)C_ * C_ * 2);
  bf16* Wr0     = (bf16*)carve((size_t)MODES_ * C_ * C_ * 2);
  bf16* Wi0     = (bf16*)carve((size_t)MODES_ * C_ * C_ * 2);
  bf16* Wr1     = (bf16*)carve((size_t)MODES_ * C_ * C_ * 2);
  bf16* Wi1     = (bf16*)carve((size_t)MODES_ * C_ * C_ * 2);
  bf16* Xr      = (bf16*)carve((size_t)MODES_ * B_ * C_ * 2);
  bf16* Xi      = (bf16*)carve((size_t)MODES_ * B_ * C_ * 2);
  bf16* Y       = (bf16*)carve((size_t)B_ * C_ * NF * 2);
  float* g      = (float*)carve((size_t)B_ * C_ * 4);

  // --- one-time prep (replayed deterministically; cheap) ---
  k_basis<<<L_ / 256, 256, 0, stream>>>(Ft, Glm);
  k_cvt_dense<<<C_ * C_ / 256, 256, 0, stream>>>(w0_w, w0bf);
  k_cvt_dense<<<C_ * C_ / 256, 256, 0, stream>>>(w1_w, w1bf);
  k_cvt_spec<<<MODES_ * C_ * C_ / 256, 256, 0, stream>>>(sw0_r, Wr0);
  k_cvt_spec<<<MODES_ * C_ * C_ / 256, 256, 0, stream>>>(sw0_i, Wi0);
  k_cvt_spec<<<MODES_ * C_ * C_ / 256, 256, 0, stream>>>(sw1_r, Wr1);
  k_cvt_spec<<<MODES_ * C_ * C_ / 256, 256, 0, stream>>>(sw1_i, Wi1);

  // --- lift ---
  k_fc0<<<(B_ * L_ * C_) / 256, 256, 0, stream>>>(x, fc0_w, fc0_b, h_cl, h_lc);

  // --- FNO block 1 (full output needed) ---
  k_dft<<<(B_ * C_ / 32) / 4, 128, 0, stream>>>(h_cl, Ft, Xr, Xi);
  k_modemix<<<(MODES_ * 8 * 16) / 4, 128, 0, stream>>>(Xr, Xi, Wr0, Wi0, Y);
  k_fno_block<4, 4><<<dim3(B_, 16, L_ / 256), 128, 0, stream>>>(
      h_lc, L_, 0, Y, w0bf, w0_b, Glm, /*l_base=*/0,
      h1_cl, h1_tail, nullptr);

  // --- FNO block 2 (only last column of output is used downstream) ---
  k_dft<<<(B_ * C_ / 32) / 4, 128, 0, stream>>>(h1_cl, Ft, Xr, Xi);
  k_modemix<<<(MODES_ * 8 * 16) / 4, 128, 0, stream>>>(Xr, Xi, Wr1, Wi1, Y);
  k_fno_block<1, 1><<<dim3(B_, 16, 1), 32, 0, stream>>>(
      h1_tail, 16, L_ - 16, Y, w1bf, w1_b, Glm, /*l_base=*/L_ - 16,
      nullptr, nullptr, g);

  // --- head ---
  k_head<<<B_, FC1N, 0, stream>>>(g, fc1_w, fc1_b, fc2_w, fc2_b, out);
}
